// CustomLoss_41970420418109
// MI455X (gfx1250) — compile-verified, matrix operand order checked by reading, not run on hardware
//
#include <hip/hip_runtime.h>
#include <math.h>
#include <float.h>

typedef float    v8f  __attribute__((ext_vector_type(8)));
typedef float    v2f  __attribute__((ext_vector_type(2)));
typedef _Float16 v16h __attribute__((ext_vector_type(16)));
typedef __fp16   pk2  __attribute__((ext_vector_type(2)));  // cvt_pkrtz result type

#define NBLOCKS 512
#define TPB 256
#define WPB (TPB / 32)   // waves per block (wave32)

// ReLU via med3 with finite upper bound (resists maxnum canonicalize rewrite)
__device__ __forceinline__ float relu1(float x) {
  return __builtin_amdgcn_fmed3f(x, 0.0f, FLT_MAX);
}

// Packed f16 ReLU on a dword holding two halves: single VOP3P instruction.
__device__ __forceinline__ int pk_relu_f16x2(int x) {
  int r;
  asm("v_pk_max_num_f16 %0, %1, 0" : "=v"(r) : "v"(x));
  return r;
}

// ---------------------------------------------------------------------------
// Main kernel: per wave, grid-stride over tiles of 16 states.
//   Layer 1:  h1^T = W1^T (64x4) @ X^T (4x16)   -> 4x v_wmma_f32_16x16x4_f32
//   Layer 2:  h2^T = W2^T (64x64) @ h1^T (64x16)-> 8x v_wmma_f32_16x16x32_f16
//   Layer 3 + sum(logits): folded into 32 FMAs with W3 (loss needs only the sum)
// D-layout of layer-1 output (lane = state, VGPR = feature) matches the f16
// B-operand layout for layer 2, so no transpose / LDS traffic in the loop.
// ---------------------------------------------------------------------------
__global__ __launch_bounds__(TPB) void mlp_sum_kernel(
    const float* __restrict__ states,
    const float* __restrict__ W1, const float* __restrict__ b1,
    const float* __restrict__ W2, const float* __restrict__ b2,
    const float* __restrict__ W3,
    float* __restrict__ partials, int nTiles)
{
  __shared__ __align__(16) _Float16 w2t[64 * 66]; // W2^T, f16, padded stride 66
  __shared__ float sWave[WPB];

  const int tid = threadIdx.x;

  // Stage W2^T into LDS (coalesced global read, transposed f16 write).
  #pragma unroll 4
  for (int idx = tid; idx < 64 * 64; idx += TPB) {
    int k = idx >> 6, n = idx & 63;
    w2t[n * 66 + k] = (_Float16)W2[idx];          // W2[k][n] -> w2t[n][k]
  }

  const int lane = tid & 31;
  const int wave = tid >> 5;
  const int l16  = lane & 15;
  const int hi   = lane >> 4;   // 0: lanes 0-15, 1: lanes 16-31
  const int hi8  = hi * 8;

  // ---- wave-resident constants ----
  // Layer-1 A operand (f32 16x4): lanes0-15 hold K=0,1 ; lanes16-31 hold K=2,3
  v2f a1w[4];
  #pragma unroll
  for (int f = 0; f < 4; ++f) {
    int col = f * 16 + l16;
    a1w[f][0] = W1[(hi ? 2 : 0) * 64 + col];
    a1w[f][1] = W1[(hi ? 3 : 1) * 64 + col];
  }
  // Bias / W3 broadcast tiles in C/D layout (VGPR r <-> feature f*16 + r + hi8)
  v8f b1c[4], b2c[4], w3c[4];
  #pragma unroll
  for (int f = 0; f < 4; ++f)
    #pragma unroll
    for (int r = 0; r < 8; ++r) {
      b1c[f][r] = b1[f * 16 + r + hi8];
      b2c[f][r] = b2[f * 16 + r + hi8];
      w3c[f][r] = W3[f * 16 + r + hi8];
    }

  __syncthreads();  // w2t ready

  // Layer-2 A operands: W2^T as 4 M-tiles x 2 K-chunks of 16x32 f16.
  // 16-bit A layout: lanes0-15: V0..V3 = K 0..7, V4..V7 = K 16..23 (pairs);
  //                  lanes16-31: K 8..15 / 24..31.
  v16h a2[4][2];
  #pragma unroll
  for (int m = 0; m < 4; ++m)
    #pragma unroll
    for (int c = 0; c < 2; ++c) {
      int Mrow = m * 16 + l16;
      int kb   = c * 32 + hi8;
      #pragma unroll
      for (int v = 0; v < 8; ++v) {
        int koff = (v < 4) ? (kb + 2 * v) : (kb + 16 + 2 * (v - 4));
        pk2 p = *reinterpret_cast<const pk2*>(&w2t[Mrow * 66 + koff]);
        a2[m][c][2 * v]     = (_Float16)p[0];
        a2[m][c][2 * v + 1] = (_Float16)p[1];
      }
    }

  // ---- grid-stride over 16-state tiles, software-pipelined X load ----
  const int gwave  = blockIdx.x * WPB + wave;
  const int nwaves = gridDim.x * WPB;
  const int xoff   = l16 * 4 + hi * 2;     // per-lane element offset in a tile
  v8f accv = {0.f, 0.f, 0.f, 0.f, 0.f, 0.f, 0.f, 0.f};

  v2f bx = {0.f, 0.f};
  if (gwave < nTiles)
    bx = *reinterpret_cast<const v2f*>(states + (size_t)gwave * 64 + xoff);

  for (int t = gwave; t < nTiles; t += nwaves) {
    // Issue next tile's load early (condition is wave-uniform)
    const int tn = t + nwaves;
    v2f bxn = bx;
    if (tn < nTiles) {
      bxn = *reinterpret_cast<const v2f*>(states + (size_t)tn * 64 + xoff);
      __builtin_prefetch(states + (size_t)tn * 64 + 4096, 0, 0);
    }

    // Layer 1 (exact f32 WMMA), bias via C input.
    v8f d1[4];
    #pragma unroll
    for (int f = 0; f < 4; ++f)
      d1[f] = __builtin_amdgcn_wmma_f32_16x16x4_f32(
                  false, a1w[f], false, bx, (short)0, b1c[f], false, false);

    // Pack f32->f16 pairs into layer-2 B layout, then packed ReLU
    // (cvt is monotonic so relu-after-cvt == relu-before-cvt).
    // Chunk c: V0..V3 <- tile 2c VGPRs (2v,2v+1); V4..V7 <- tile 2c+1.
    v16h bh[2];
    #pragma unroll
    for (int c = 0; c < 2; ++c) {
      struct I8 { int w[8]; } s;
      #pragma unroll
      for (int j = 0; j < 4; ++j) {
        pk2 q0 = __builtin_amdgcn_cvt_pkrtz(d1[2 * c][2 * j],
                                            d1[2 * c][2 * j + 1]);
        pk2 q1 = __builtin_amdgcn_cvt_pkrtz(d1[2 * c + 1][2 * j],
                                            d1[2 * c + 1][2 * j + 1]);
        s.w[j]     = pk_relu_f16x2(__builtin_bit_cast(int, q0));
        s.w[4 + j] = pk_relu_f16x2(__builtin_bit_cast(int, q1));
      }
      bh[c] = __builtin_bit_cast(v16h, s);
    }

    // Layer 2 (f16 WMMA, bias via C) + fold layer-3 dot with W3 into 8
    // independent accumulator lanes (short dependency chains).
    #pragma unroll
    for (int m = 0; m < 4; ++m) {
      v8f d2 = __builtin_amdgcn_wmma_f32_16x16x32_f16(
                   false, a2[m][0], false, bh[0], (short)0, b2c[m], false, false);
      d2 = __builtin_amdgcn_wmma_f32_16x16x32_f16(
                   false, a2[m][1], false, bh[1], (short)0, d2, false, false);
      #pragma unroll
      for (int r = 0; r < 8; ++r)
        accv[r] = fmaf(relu1(d2[r]), w3c[m][r], accv[r]);
    }

    bx = bxn;
  }

  // Collapse the 8 accumulator lanes, then deterministic block reduction.
  float acc = (((accv[0] + accv[1]) + (accv[2] + accv[3])) +
               ((accv[4] + accv[5]) + (accv[6] + accv[7])));
  #pragma unroll
  for (int off = 16; off > 0; off >>= 1)
    acc += __shfl_xor(acc, off, 32);
  if (lane == 0) sWave[wave] = acc;
  __syncthreads();
  if (tid == 0) {
    float s = 0.f;
    #pragma unroll
    for (int w = 0; w < WPB; ++w) s += sWave[w];
    partials[blockIdx.x] = s;
  }
}

// ---------------------------------------------------------------------------
// Finalize: denominator path net(eye(4)) in f32/f64, logsumexp, combine.
// loss = -( sum_logits + N*b3 - N*logsumexp(net(e_i)) )
// ---------------------------------------------------------------------------
__global__ void finalize_kernel(
    const float* __restrict__ W1, const float* __restrict__ b1,
    const float* __restrict__ W2, const float* __restrict__ b2,
    const float* __restrict__ W3, const float* __restrict__ b3,
    const float* __restrict__ partials, int nBlocks, long long Nstates,
    float* __restrict__ out)
{
  __shared__ float  h1s[4][64];
  __shared__ float  h2s[4][64];
  __shared__ double red[64];
  const int j = threadIdx.x;  // 64 threads

  #pragma unroll
  for (int e = 0; e < 4; ++e)
    h1s[e][j] = fmaxf(W1[e * 64 + j] + b1[j], 0.f);   // x = e_e -> row of W1
  __syncthreads();

  #pragma unroll
  for (int e = 0; e < 4; ++e) {
    float a = b2[j];
    for (int k = 0; k < 64; ++k)
      a = fmaf(h1s[e][k], W2[k * 64 + j], a);
    h2s[e][j] = fmaxf(a, 0.f);
  }

  double ps = 0.0;
  for (int i = j; i < nBlocks; i += 64) ps += (double)partials[i];
  red[j] = ps;
  __syncthreads();

  if (j == 0) {
    double S = 0.0;
    for (int w = 0; w < 64; ++w) S += red[w];
    double l[4], mx = -1e300;
    for (int e = 0; e < 4; ++e) {
      double v = (double)b3[0];
      for (int k = 0; k < 64; ++k) v += (double)h2s[e][k] * (double)W3[k];
      l[e] = v;
      if (v > mx) mx = v;
    }
    double se = 0.0;
    for (int e = 0; e < 4; ++e) se += exp(l[e] - mx);
    double logden = mx + log(se);
    double Nd = (double)Nstates;
    out[0] = (float)(-(S + Nd * (double)b3[0] - Nd * logden));
  }
}

extern "C" void kernel_launch(void* const* d_in, const int* in_sizes, int n_in,
                              void* d_out, int out_size, void* d_ws, size_t ws_size,
                              hipStream_t stream) {
  const float* states = (const float*)d_in[0];
  const float* W1 = (const float*)d_in[1];
  const float* b1 = (const float*)d_in[2];
  const float* W2 = (const float*)d_in[3];
  const float* b2 = (const float*)d_in[4];
  const float* W3 = (const float*)d_in[5];
  const float* b3 = (const float*)d_in[6];

  const int Ntot   = in_sizes[0] / 4;   // 2,097,152 states
  const int nTiles = Ntot / 16;
  float* partials = (float*)d_ws;       // NBLOCKS floats

  mlp_sum_kernel<<<NBLOCKS, TPB, 0, stream>>>(states, W1, b1, W2, b2, W3,
                                              partials, nTiles);
  finalize_kernel<<<1, 64, 0, stream>>>(W1, b1, W2, b2, W3, b3,
                                        partials, NBLOCKS, (long long)Ntot,
                                        (float*)d_out);
}
